// AdvancedAttn_71871982731692
// MI455X (gfx1250) — matrix operand order repaired and checked by reading, not compiled, not run on hardware
//
#include <hip/hip_runtime.h>
#include <hip/hip_bf16.h>
#include <stdint.h>

typedef __bf16 bf16_t;
typedef __attribute__((ext_vector_type(16))) __bf16 v16bf;
typedef __attribute__((ext_vector_type(8)))  float  v8f;

#define WMMA_BF16(a, b, c) \
  __builtin_amdgcn_wmma_f32_16x16x32_bf16(false, (a), false, (b), (short)0, (c), false, false)

// CDNA5 16-bit A-operand (16x32) K index for packed dword g (0..7), lane-half hf.
static __device__ __forceinline__ int ka_map(int g, int hf) {
  return (g < 4) ? (hf * 8 + 2 * g) : (16 + hf * 8 + 2 * (g - 4));
}
// CDNA5 16-bit B-operand (32x16) K index for packed dword g, lane-half hf.
static __device__ __forceinline__ int kb_map(int g, int hf) {
  return (g < 4) ? (hf * 8 + 2 * g) : (16 + hf * 8 + 2 * (g - 4));
}

static __device__ __forceinline__ void wait_async0() {
#if __has_builtin(__builtin_amdgcn_s_wait_asynccnt)
  __builtin_amdgcn_s_wait_asynccnt(0);
#else
  asm volatile("s_wait_asynccnt 0" ::: "memory");
#endif
}

// Async DMA of a 2D bf16 tile (rows of (1<<RB_LOG2) bytes, global row stride
// `gstride` bytes) into LDS at byte offset lds0, linear LDS layout.
// Each thread moves NITER x 16 bytes; STEP = blockDim.x * 16.
template <int NITER, int STEP, int RB_LOG2>
static __device__ __forceinline__ void async_tile(const bf16_t* g, uint32_t lds0,
                                                  int t, uint32_t gstride) {
  const uint64_t gb = (uint64_t)g;   // uniform (block-level) base -> SGPR pair
#pragma unroll
  for (int r = 0; r < NITER; ++r) {
    const uint32_t o    = (uint32_t)t * 16u + (uint32_t)(r * STEP);
    const uint32_t row  = o >> RB_LOG2;
    const uint32_t cb   = o & ((1u << RB_LOG2) - 1u);
    const uint32_t goff = row * gstride + cb;  // 32-bit byte offset (GVS mode)
    const uint32_t loff = lds0 + o;            // LDS byte address
    asm volatile("global_load_async_to_lds_b128 %0, %1, %2"
                 :
                 : "v"(loff), "v"(goff), "s"(gb)
                 : "memory");
  }
}

// ---------------------------------------------------------------------------
// fp32 -> bf16 bulk convert (float4 per thread).
// ---------------------------------------------------------------------------
__global__ __launch_bounds__(256) void f32_to_bf16(const float* __restrict__ in,
                                                   bf16_t* __restrict__ out, int n4) {
  const int idx = blockIdx.x * 256 + threadIdx.x;
  if (idx >= n4) return;
  const float4 f = reinterpret_cast<const float4*>(in)[idx];
  bf16_t* o = out + (size_t)idx * 4;
  o[0] = (bf16_t)f.x; o[1] = (bf16_t)f.y; o[2] = (bf16_t)f.z; o[3] = (bf16_t)f.w;
}

// ---------------------------------------------------------------------------
// GEMM: C[M,N] = A[M,K] * W[N,K]^T, all-bf16 operands, fp32 WMMA accumulate.
// Block 256 threads (8 waves, 2x4), tile 128x128, BK=64, double-buffered LDS
// filled by global_load_async_to_lds_b128 (stage i+1 overlaps compute of i).
// ---------------------------------------------------------------------------
template <bool OUT_F32>
__global__ __launch_bounds__(256) void gemm_bf16(const bf16_t* __restrict__ A,
                                                 const bf16_t* __restrict__ W,
                                                 void* __restrict__ Cout,
                                                 int M, int N, int K) {
  constexpr int BM = 128, BN = 128, BK = 64;
  __shared__ bf16_t sA[2][BM * BK];
  __shared__ bf16_t sB[2][BN * BK];

  const int t    = threadIdx.x;
  const int lane = t & 31, wave = t >> 5;
  const int hf   = lane >> 4, lr = lane & 15;
  const int wm   = wave >> 2, wn = wave & 3;
  const int m0   = blockIdx.y * BM, n0 = blockIdx.x * BN;

  const bf16_t* Ab = A + (size_t)m0 * K;
  const bf16_t* Wb = W + (size_t)n0 * K;
  const uint32_t sAo[2] = {(uint32_t)(uintptr_t)&sA[0][0], (uint32_t)(uintptr_t)&sA[1][0]};
  const uint32_t sBo[2] = {(uint32_t)(uintptr_t)&sB[0][0], (uint32_t)(uintptr_t)&sB[1][0]};
  const uint32_t kstride = (uint32_t)(K * 2);

  v8f acc[4][2];
#pragma unroll
  for (int a = 0; a < 4; ++a)
#pragma unroll
    for (int b2 = 0; b2 < 2; ++b2)
#pragma unroll
      for (int e = 0; e < 8; ++e) acc[a][b2][e] = 0.0f;

  // Stage 0 DMA.
  async_tile<4, 4096, 7>(Ab, sAo[0], t, kstride);
  async_tile<4, 4096, 7>(Wb, sBo[0], t, kstride);

  const int nk = K / BK;
  for (int i = 0; i < nk; ++i) {
    wait_async0();
    __syncthreads();
    if (i + 1 < nk) {  // kick off next stage into the other buffer
      async_tile<4, 4096, 7>(Ab + (i + 1) * BK, sAo[(i + 1) & 1], t, kstride);
      async_tile<4, 4096, 7>(Wb + (i + 1) * BK, sBo[(i + 1) & 1], t, kstride);
    }
    const bf16_t* la = sA[i & 1];
    const bf16_t* lb = sB[i & 1];
#pragma unroll
    for (int ks = 0; ks < 2; ++ks) {
      v16bf bfr[2];
#pragma unroll
      for (int nt = 0; nt < 2; ++nt) {
        uint32_t* p = reinterpret_cast<uint32_t*>(&bfr[nt]);
        const int rb = wn * 32 + nt * 16 + lr;
#pragma unroll
        for (int g = 0; g < 8; ++g)
          p[g] = *reinterpret_cast<const uint32_t*>(&lb[rb * BK + ks * 32 + kb_map(g, hf)]);
      }
#pragma unroll
      for (int mt = 0; mt < 4; ++mt) {
        v16bf af;
        uint32_t* p = reinterpret_cast<uint32_t*>(&af);
        const int ra = wm * 64 + mt * 16 + lr;
#pragma unroll
        for (int g = 0; g < 8; ++g)
          p[g] = *reinterpret_cast<const uint32_t*>(&la[ra * BK + ks * 32 + ka_map(g, hf)]);
#pragma unroll
        for (int nt = 0; nt < 2; ++nt) acc[mt][nt] = WMMA_BF16(af, bfr[nt], acc[mt][nt]);
      }
    }
  }

#pragma unroll
  for (int mt = 0; mt < 4; ++mt)
#pragma unroll
    for (int nt = 0; nt < 2; ++nt)
#pragma unroll
      for (int j = 0; j < 8; ++j) {
        const int row = m0 + wm * 64 + mt * 16 + j + hf * 8;
        const int col = n0 + wn * 32 + nt * 16 + lr;
        if (OUT_F32)
          ((float*)Cout)[(size_t)row * N + col] = acc[mt][nt][j];
        else
          ((bf16_t*)Cout)[(size_t)row * N + col] = (bf16_t)acc[mt][nt][j];
      }
}

// ---------------------------------------------------------------------------
// RoPE in-place on [B, S, nh, 128] bf16, fp32 math.
// ---------------------------------------------------------------------------
__global__ __launch_bounds__(256) void rope_inplace(bf16_t* __restrict__ tp,
                                                    int S, int nh, int total) {
  const int idx = blockIdx.x * 256 + threadIdx.x;
  if (idx >= total) return;
  const int d = idx & 63;
  int tmp = idx >> 6;
  const int h = tmp % nh; tmp /= nh;
  const int s = tmp % S;
  const int b = tmp / S;
  const size_t base = (((size_t)b * S + s) * nh + h) * 128;
  const float x1 = (float)tp[base + d];
  const float x2 = (float)tp[base + d + 64];
  const float ang = (float)s * exp2f(-(float)d * 0.20762050593045935f);
  float sn, cs;
  __sincosf(ang, &sn, &cs);
  tp[base + d]      = (bf16_t)(x1 * cs - x2 * sn);
  tp[base + d + 64] = (bf16_t)(x2 * cs + x1 * sn);
}

// ---------------------------------------------------------------------------
// V transpose: [B,S,KV,128] -> [B,KV,128,S]
// ---------------------------------------------------------------------------
__global__ __launch_bounds__(256) void transpose_v(const bf16_t* __restrict__ v,
                                                   bf16_t* __restrict__ vt,
                                                   int S, int total) {
  const int idx = blockIdx.x * 256 + threadIdx.x;
  if (idx >= total) return;
  const int s = idx % S;
  int tmp = idx / S;
  const int d = tmp & 127; tmp >>= 7;
  const int g = tmp & 3;
  const int b = tmp >> 2;
  vt[idx] = v[(((size_t)b * S + s) * 4 + g) * 128 + d];
}

// ---------------------------------------------------------------------------
// Flash attention, GQA (H=16, KV=4, HD=128), causal, online softmax.
// Block = 256 threads = 8 waves; 128 query rows per block (16 per wave).
// K (64x128) and Vt (128x64) tiles are double-buffered in LDS, filled with
// async DMA shared by all 8 waves; next key block overlaps current compute.
// Causal structure: waves skip fully-masked key blocks (wave-uniform guard,
// barriers stay unconditional) and only the diagonal block applies the mask.
// ---------------------------------------------------------------------------
__global__ __launch_bounds__(256) void flash_gqa(const bf16_t* __restrict__ Qm,
                                                 const bf16_t* __restrict__ Km,
                                                 const bf16_t* __restrict__ Vt,
                                                 bf16_t* __restrict__ Om, int S) {
  const int qt = blockIdx.x, h = blockIdx.y, b = blockIdx.z;
  const int gh = h >> 2;
  const int t = threadIdx.x, lane = t & 31, wave = t >> 5;
  const int hf = lane >> 4, lr = lane & 15;
  const int q0 = qt * 128 + wave * 16;
  const int idiag = q0 >> 6;          // key-block index containing the diagonal

  __shared__ bf16_t sK[2][64 * 128];  // [key_local][d]
  __shared__ bf16_t sV[2][128 * 64];  // [d][key_local]
  __shared__ bf16_t sP[8][16 * 64];   // wave-private P re-layout

  const uint32_t sKo[2] = {(uint32_t)(uintptr_t)&sK[0][0], (uint32_t)(uintptr_t)&sK[1][0]};
  const uint32_t sVo[2] = {(uint32_t)(uintptr_t)&sV[0][0], (uint32_t)(uintptr_t)&sV[1][0]};

  const bf16_t* Kh = Km + (size_t)gh * 128;                       // + (b*S+key)*512
  const bf16_t* Vh = Vt + ((size_t)b * 4 + gh) * 128 * (size_t)S; // row d stride S

  // Preload Q A-fragments for the 4 HD chunks of 32.
  v16bf qf[4];
  {
    const bf16_t* qrow = Qm + (((size_t)b * S + (q0 + lr)) * 16 + h) * 128;
#pragma unroll
    for (int ks = 0; ks < 4; ++ks) {
      uint32_t* p = reinterpret_cast<uint32_t*>(&qf[ks]);
#pragma unroll
      for (int g = 0; g < 8; ++g)
        p[g] = *reinterpret_cast<const uint32_t*>(qrow + ks * 32 + ka_map(g, hf));
    }
  }

  float mrow[8], lsum[8];
  v8f oacc[8];
#pragma unroll
  for (int j = 0; j < 8; ++j) { mrow[j] = -3.0e38f; lsum[j] = 0.f; }
#pragma unroll
  for (int dt = 0; dt < 8; ++dt)
#pragma unroll
    for (int j = 0; j < 8; ++j) oacc[dt][j] = 0.f;

  const float scale = 0.088388347648318447f;  // 1/sqrt(128)

  // Stage 0 DMA: K rows 256B at global stride 1024B; V rows 128B at stride S*2.
  async_tile<4, 4096, 8>(Kh + (size_t)b * S * 512, sKo[0], t, 1024u);
  async_tile<4, 4096, 7>(Vh, sVo[0], t, (uint32_t)(S * 2));

  const int nblk = 2 * qt + 2;  // key blocks covering rows [0, qt*128+127]
  for (int i = 0; i < nblk; ++i) {
    const int kb = i * 64;
    wait_async0();
    __syncthreads();
    if (i + 1 < nblk) {
      async_tile<4, 4096, 8>(Kh + ((size_t)b * S + kb + 64) * 512, sKo[(i + 1) & 1], t, 1024u);
      async_tile<4, 4096, 7>(Vh + (kb + 64), sVo[(i + 1) & 1], t, (uint32_t)(S * 2));
    }
    const bf16_t* lk = sK[i & 1];
    const bf16_t* lv = sV[i & 1];
    const bool active = (i <= idiag);  // wave-uniform
    const bool diag   = (i == idiag);  // wave-uniform

    if (active) {
      // ---- scores = Q . K^T for 64 keys (4 tiles of 16) ----
      v8f sacc[4];
#pragma unroll
      for (int nt = 0; nt < 4; ++nt)
#pragma unroll
        for (int j = 0; j < 8; ++j) sacc[nt][j] = 0.f;

#pragma unroll
      for (int nt = 0; nt < 4; ++nt) {
        const int krow = (nt * 16 + lr) * 128;
#pragma unroll
        for (int ks = 0; ks < 4; ++ks) {
          v16bf kf;
          uint32_t* p = reinterpret_cast<uint32_t*>(&kf);
#pragma unroll
          for (int g = 0; g < 8; ++g)
            p[g] = *reinterpret_cast<const uint32_t*>(&lk[krow + ks * 32 + kb_map(g, hf)]);
          sacc[nt] = WMMA_BF16(qf[ks], kf, sacc[nt]);
        }
      }

      // ---- scale (+ causal mask on diagonal block only) + online softmax ----
#pragma unroll
      for (int j = 0; j < 8; ++j) {
        const int qi = q0 + j + hf * 8;
        float sv[4];
        float mx = -3.0e38f;
        if (diag) {
#pragma unroll
          for (int nt = 0; nt < 4; ++nt) {
            const int key = kb + nt * 16 + lr;
            const float s = sacc[nt][j] * scale + (key > qi ? -1.0e9f : 0.0f);
            sv[nt] = s;
            mx = fmaxf(mx, s);
          }
        } else {
#pragma unroll
          for (int nt = 0; nt < 4; ++nt) {
            const float s = sacc[nt][j] * scale;
            sv[nt] = s;
            mx = fmaxf(mx, s);
          }
        }
#pragma unroll
        for (int d = 1; d < 16; d <<= 1) mx = fmaxf(mx, __shfl_xor(mx, d, 32));
        const float mn = fmaxf(mrow[j], mx);
        const float sf = __expf(mrow[j] - mn);
        float rs = 0.f;
#pragma unroll
        for (int nt = 0; nt < 4; ++nt) {
          const float pv = __expf(sv[nt] - mn);
          rs += pv;
          sP[wave][(j + hf * 8) * 64 + nt * 16 + lr] = (bf16_t)pv;
        }
#pragma unroll
        for (int d = 1; d < 16; d <<= 1) rs += __shfl_xor(rs, d, 32);
        lsum[j] = lsum[j] * sf + rs;
#pragma unroll
        for (int dt = 0; dt < 8; ++dt) oacc[dt][j] *= sf;
        mrow[j] = mn;
      }
    }
    __syncthreads();  // unconditional: sP store -> sP read (and pipeline safety)

    if (active) {
      // ---- O += P . V (2 WMMA k-steps over 64 keys) ----
#pragma unroll
      for (int s2 = 0; s2 < 2; ++s2) {
        v16bf pf;
        uint32_t* p = reinterpret_cast<uint32_t*>(&pf);
#pragma unroll
        for (int g = 0; g < 8; ++g)
          p[g] = *reinterpret_cast<const uint32_t*>(
              &sP[wave][lr * 64 + s2 * 32 + ka_map(g, hf)]);
#pragma unroll
        for (int dt = 0; dt < 8; ++dt) {
          v16bf vf;
          uint32_t* pv2 = reinterpret_cast<uint32_t*>(&vf);
          const int vrow = (dt * 16 + lr) * 64;
#pragma unroll
          for (int g = 0; g < 8; ++g)
            pv2[g] = *reinterpret_cast<const uint32_t*>(&lv[vrow + s2 * 32 + kb_map(g, hf)]);
          oacc[dt] = WMMA_BF16(pf, vf, oacc[dt]);
        }
      }
    }
  }

  // ---- normalize and store O in [B,S,16,128] ----
#pragma unroll
  for (int dt = 0; dt < 8; ++dt)
#pragma unroll
    for (int j = 0; j < 8; ++j) {
      const int qi = q0 + j + hf * 8;
      const float val = oacc[dt][j] / lsum[j];
      Om[(((size_t)b * S + qi) * 16 + h) * 128 + dt * 16 + lr] = (bf16_t)val;
    }
}

// ---------------------------------------------------------------------------
extern "C" void kernel_launch(void* const* d_in, const int* in_sizes, int n_in,
                              void* d_out, int out_size, void* d_ws, size_t ws_size,
                              hipStream_t stream) {
  (void)in_sizes; (void)n_in; (void)out_size; (void)ws_size;
  const float* x  = (const float*)d_in[0];
  const float* wq = (const float*)d_in[1];
  const float* wk = (const float*)d_in[2];
  const float* wv = (const float*)d_in[3];
  const float* wo = (const float*)d_in[4];

  constexpr int B = 2, S = 2048, D = 2048, H = 16, KV = 4, HD = 128;
  const int M = B * S;  // 4096

  char* ws = (char*)d_ws;
  size_t off = 0;
  auto alloc = [&](size_t elems) { bf16_t* p = (bf16_t*)(ws + off); off += elems * 2; return p; };
  bf16_t* xb   = alloc((size_t)M * D);
  bf16_t* wqb  = alloc((size_t)D * D);
  bf16_t* wkb  = alloc((size_t)KV * HD * D);
  bf16_t* wvb  = alloc((size_t)KV * HD * D);
  bf16_t* wob  = alloc((size_t)D * D);
  bf16_t* qbuf = alloc((size_t)M * D);
  bf16_t* kbuf = alloc((size_t)M * KV * HD);
  bf16_t* vbuf = alloc((size_t)M * KV * HD);
  bf16_t* vtb  = alloc((size_t)M * KV * HD);
  bf16_t* obuf = alloc((size_t)M * D);

  // fp32 -> bf16 conversions (one pass; makes GEMM tiles pure byte-moves).
  f32_to_bf16<<<(M * D / 4) / 256, 256, 0, stream>>>(x, xb, M * D / 4);
  f32_to_bf16<<<(D * D / 4) / 256, 256, 0, stream>>>(wq, wqb, D * D / 4);
  f32_to_bf16<<<(KV * HD * D / 4) / 256, 256, 0, stream>>>(wk, wkb, KV * HD * D / 4);
  f32_to_bf16<<<(KV * HD * D / 4) / 256, 256, 0, stream>>>(wv, wvb, KV * HD * D / 4);
  f32_to_bf16<<<(D * D / 4) / 256, 256, 0, stream>>>(wo, wob, D * D / 4);

  // Projections.
  gemm_bf16<false><<<dim3(D / 128, M / 128), 256, 0, stream>>>(xb, wqb, qbuf, M, D, D);
  gemm_bf16<false><<<dim3((KV * HD) / 128, M / 128), 256, 0, stream>>>(xb, wkb, kbuf, M, KV * HD, D);
  gemm_bf16<false><<<dim3((KV * HD) / 128, M / 128), 256, 0, stream>>>(xb, wvb, vbuf, M, KV * HD, D);

  // RoPE (in-place) and V transpose.
  rope_inplace<<<(M * H * 64) / 256, 256, 0, stream>>>(qbuf, S, H, M * H * 64);
  rope_inplace<<<(M * KV * 64) / 256, 256, 0, stream>>>(kbuf, S, KV, M * KV * 64);
  transpose_v<<<(M * KV * HD) / 256, 256, 0, stream>>>(vbuf, vtb, S, M * KV * HD);

  // Causal GQA flash attention (128 q rows per block, 8 waves).
  flash_gqa<<<dim3(S / 128, H, B), 256, 0, stream>>>(qbuf, kbuf, vtb, obuf, S);

  // Output projection (fp32 out into d_out).
  gemm_bf16<true><<<dim3(D / 128, M / 128), 256, 0, stream>>>(obuf, wob, d_out, M, D, D);
}
